// RNN_85134841741958
// MI455X (gfx1250) — matrix operand order repaired and checked by reading, not compiled, not run on hardware
//
#include <hip/hip_runtime.h>
#include <hip/hip_bf16.h>
#include <stdint.h>

// Problem constants (match reference)
#define HD 1024
#define VD 50257
#define TD 256
#define NWG_RNN 64          // workgroups cooperating on the recurrence
#define ROWS_PER_WG 16      // 64 * 16 = 1024 rows of h

// GEMM tiling
#define BN   128            // columns per block
#define BK   32             // K-chunk staged in LDS
#define NBLK 393            // ceil(50257 / 128)
#define AST  40             // LDS A row stride in ushorts (32 elems + 8 pad = 80B)
#define BST  40             // LDS B col stride in ushorts

typedef __attribute__((ext_vector_type(16))) __bf16 v16bf;
typedef __attribute__((ext_vector_type(8)))  float  v8f;

__device__ __forceinline__ unsigned short f32_bf16(float f) {
    unsigned int u = __float_as_uint(f);
    unsigned int r = u + 0x7FFFu + ((u >> 16) & 1u);   // round-to-nearest-even
    return (unsigned short)(r >> 16);
}

// ---------------- Phase 0: init workspace (every launch; graph-safe) --------
__global__ void RNN_init_ws(int* bar, float* hbuf, const float* h0) {
    int tid = threadIdx.x;
    if (tid < TD) bar[tid] = 0;
    for (int i = tid; i < HD; i += 1024) hbuf[i] = h0[i];  // ping buffer 0 = h0
}

// ---------------- Phase 1: sequential recurrence over 64 workgroups --------
__global__ void RNN_steps(const int* __restrict__ chars,
                          const float* __restrict__ Wxh,
                          const float* __restrict__ Whh,
                          const float* __restrict__ bh,
                          int* bar, float* hbuf,
                          unsigned short* __restrict__ Hb) {
    __shared__ float hprev[HD];
    __shared__ float part[256];
    const int tid  = threadIdx.x;
    const int wg   = blockIdx.x;
    const int seg  = tid & 15;                 // 16 K-segments of 64
    const int row  = wg * ROWS_PER_WG + (tid >> 4);
    const float4* w4 = (const float4*)(Whh + (size_t)row * HD + seg * 64);

    for (int t = 0; t < TD; ++t) {
        const float* hsrc = hbuf + (t & 1) * HD;
        float*       hdst = hbuf + ((t + 1) & 1) * HD;
        for (int i = tid; i < HD; i += 256) hprev[i] = hsrc[i];
        __syncthreads();

        float acc = 0.f;
        #pragma unroll
        for (int i = 0; i < 16; ++i) {
            float4 w = w4[i];
            int k = seg * 64 + i * 4;
            acc += w.x * hprev[k] + w.y * hprev[k + 1]
                 + w.z * hprev[k + 2] + w.w * hprev[k + 3];
        }
        part[tid] = acc;
        __syncthreads();

        if (tid < ROWS_PER_WG) {
            int r = wg * ROWS_PER_WG + tid;
            float s = 0.f;
            #pragma unroll
            for (int i = 0; i < 16; ++i) s += part[tid * 16 + i];
            int c = chars[t];
            s += Wxh[(size_t)r * VD + c] + bh[r];          // Wxh @ one_hot(c)
            float hn = tanhf(s);
            hdst[r] = hn;
            Hb[t * HD + r] = f32_bf16(hn);                 // A-matrix for GEMM
            __threadfence();
        }
        __syncthreads();
        if (tid == 0) {
            __hip_atomic_fetch_add(&bar[t], 1, __ATOMIC_RELEASE,
                                   __HIP_MEMORY_SCOPE_AGENT);
            while (__hip_atomic_load(&bar[t], __ATOMIC_ACQUIRE,
                                     __HIP_MEMORY_SCOPE_AGENT) < NWG_RNN)
                __builtin_amdgcn_s_sleep(1);
        }
        __syncthreads();
    }
}

// ---------------- Phase 2: logits = H(256x1024) @ Why^T via WMMA bf16 ------
// Block: 16 waves = 4 M-groups x 4 N-groups; wave computes 4 M-tiles x 2
// N-tiles (64x32 of C). Per BK=32 chunk, A and B are staged once into LDS
// (B converted f32->bf16 with one v_perm per pair); Why read exactly once.
__global__ void __launch_bounds__(512)
RNN_gemm_logits(const unsigned short* __restrict__ Hb,
                const float* __restrict__ Why,
                const float* __restrict__ by,
                float* __restrict__ out) {
    __shared__ __align__(16) unsigned short Alds[256 * AST]; // 20 KB
    __shared__ __align__(16) unsigned short Blds[BN  * BST]; // 10 KB

    const int tid   = threadIdx.x;
    const int lane  = tid & 31;
    const int wave  = tid >> 5;         // 0..15
    const int wm    = wave >> 2;        // rows  [wm*64, wm*64+64)
    const int wn    = wave & 3;         // cols  [wn*32, wn*32+32) within strip
    const int lhalf = lane >> 4;
    const int lmod  = lane & 15;
    const int n0    = blockIdx.x * BN;

    // staging work split
    const int arow  = tid >> 1;                 // 0..255, 16B (8 bf16) each
    const int ahalf = tid & 1;
    const int bcolS = tid >> 2;                 // 0..127, 8 f32 each
    const int bq    = tid & 3;
    int bsrc = n0 + bcolS; if (bsrc >= VD) bsrc = VD - 1;   // clamp tail (stores guarded)

    v8f acc[4][2] = {};

    for (int kc = 0; kc < HD; kc += BK) {
        // ---- stage A chunk: Hb rows 0..255, K kc..kc+31 (plain bf16 copy) ----
        {
            const uint4* src = (const uint4*)(Hb + (size_t)arow * HD + kc + ahalf * 16);
            *(uint4*)(Alds + arow * AST + ahalf * 16) = *src;
        }
        // ---- stage B chunk: cols n0..n0+127, convert f32 -> bf16 (1 perm / pair) ----
        {
            const float4* src = (const float4*)(Why + (size_t)bsrc * HD + kc + bq * 8);
            float4 f0 = src[0], f1 = src[1];
            uint4 packed;
            packed.x = __builtin_amdgcn_perm(__float_as_uint(f0.y), __float_as_uint(f0.x), 0x07060302u);
            packed.y = __builtin_amdgcn_perm(__float_as_uint(f0.w), __float_as_uint(f0.z), 0x07060302u);
            packed.z = __builtin_amdgcn_perm(__float_as_uint(f1.y), __float_as_uint(f1.x), 0x07060302u);
            packed.w = __builtin_amdgcn_perm(__float_as_uint(f1.w), __float_as_uint(f1.z), 0x07060302u);
            *(uint4*)(Blds + bcolS * BST + bq * 8) = packed;
        }
        __syncthreads();

        // ---- fragments from LDS (ds_load_b128, conflict-free via 80B stride) ----
        union { uint4 q[2]; v16bf v; } fa[4], fb[2];
        #pragma unroll
        for (int mt = 0; mt < 4; ++mt) {
            const unsigned short* p = Alds + (wm * 64 + mt * 16 + lmod) * AST + lhalf * 8;
            fa[mt].q[0] = *(const uint4*)(p);        // K = lhalf*8 + 0..7
            fa[mt].q[1] = *(const uint4*)(p + 16);   // K = lhalf*8 + 16..23
        }
        #pragma unroll
        for (int nt = 0; nt < 2; ++nt) {
            const unsigned short* p = Blds + (wn * 32 + nt * 16 + lmod) * BST + lhalf * 8;
            fb[nt].q[0] = *(const uint4*)(p);        // K = lhalf*8 + 0..7
            fb[nt].q[1] = *(const uint4*)(p + 16);   // K = lhalf*8 + 16..23
        }

        #pragma unroll
        for (int mt = 0; mt < 4; ++mt)
            #pragma unroll
            for (int nt = 0; nt < 2; ++nt)
                acc[mt][nt] = __builtin_amdgcn_wmma_f32_16x16x32_bf16(
                    false, fa[mt].v, false, fb[nt].v, (short)0, acc[mt][nt],
                    false, false);
        __syncthreads();
    }

    // ---- epilogue: D layout rows = m0 + lhalf*8 + r, col = n tile + lmod ----
    #pragma unroll
    for (int nt = 0; nt < 2; ++nt) {
        int col = n0 + wn * 32 + nt * 16 + lmod;
        if (col < VD) {
            float bias = by[col];
            #pragma unroll
            for (int mt = 0; mt < 4; ++mt) {
                int rbase = wm * 64 + mt * 16 + lhalf * 8;
                #pragma unroll
                for (int r = 0; r < 8; ++r)
                    out[(size_t)(rbase + r) * VD + col] = acc[mt][nt][r] + bias;
            }
        }
    }
}

// ---------------- Phase 3: row softmax over V (in-place in d_out) ----------
__global__ void RNN_softmax(float* __restrict__ out) {
    __shared__ float red[256];
    const int row = blockIdx.x;
    const int tid = threadIdx.x;
    float* p = out + (size_t)row * VD;

    float m = -3.4e38f;
    for (int j = tid; j < VD; j += 256) m = fmaxf(m, p[j]);
    red[tid] = m; __syncthreads();
    for (int s = 128; s > 0; s >>= 1) {
        if (tid < s) red[tid] = fmaxf(red[tid], red[tid + s]);
        __syncthreads();
    }
    m = red[0]; __syncthreads();

    float sum = 0.f;
    for (int j = tid; j < VD; j += 256) sum += __expf(p[j] - m);
    red[tid] = sum; __syncthreads();
    for (int s = 128; s > 0; s >>= 1) {
        if (tid < s) red[tid] += red[tid + s];
        __syncthreads();
    }
    float inv = 1.0f / red[0];
    __syncthreads();
    for (int j = tid; j < VD; j += 256) p[j] = __expf(p[j] - m) * inv;
}

extern "C" void kernel_launch(void* const* d_in, const int* in_sizes, int n_in,
                              void* d_out, int out_size, void* d_ws, size_t ws_size,
                              hipStream_t stream) {
    (void)in_sizes; (void)n_in; (void)out_size; (void)ws_size;
    const int*   chars = (const int*)d_in[0];
    const float* Wxh   = (const float*)d_in[1];
    const float* Whh   = (const float*)d_in[2];
    const float* Why   = (const float*)d_in[3];
    const float* bh    = (const float*)d_in[4];
    const float* by    = (const float*)d_in[5];
    const float* h0    = (const float*)d_in[6];
    float* out = (float*)d_out;

    // workspace layout (total 540,672 bytes needed):
    char* ws = (char*)d_ws;
    int*            bar  = (int*)ws;                       // [0,1024): barrier slots
    float*          hbuf = (float*)(ws + 1024);            // [1024,9216): h ping-pong
    unsigned short* Hb   = (unsigned short*)(ws + 16384);  // [16K,528K): H bf16 (256x1024)

    hipLaunchKernelGGL(RNN_init_ws, dim3(1), dim3(1024), 0, stream, bar, hbuf, h0);
    hipLaunchKernelGGL(RNN_steps, dim3(NWG_RNN), dim3(256), 0, stream,
                       chars, Wxh, Whh, bh, bar, hbuf, Hb);
    hipLaunchKernelGGL(RNN_gemm_logits, dim3(NBLK), dim3(512), 0, stream,
                       Hb, Why, by, out);
    hipLaunchKernelGGL(RNN_softmax, dim3(TD), dim3(256), 0, stream, out);
}
